// GraphVQVariationalAutoencoder_3504693314187
// MI455X (gfx1250) — matrix operand order repaired and checked by reading, not compiled, not run on hardware
//
#include <hip/hip_runtime.h>
#include <stdint.h>

// ---------------- types for CDNA5 WMMA / TDM ----------------
typedef __bf16   v16bf __attribute__((ext_vector_type(16)));
typedef float    v8f   __attribute__((ext_vector_type(8)));
typedef unsigned v4u   __attribute__((ext_vector_type(4)));
typedef int      v4i   __attribute__((ext_vector_type(4)));
typedef int      v8i   __attribute__((ext_vector_type(8)));

#if __has_builtin(__builtin_amdgcn_tensor_load_to_lds)
#define HAVE_TDM 1
#endif

// ---------------- problem constants ----------------
static constexpr int B  = 32;
static constexpr int S  = 576;
static constexpr int F  = 700;
static constexpr int D  = 16;
static constexpr int K_CB = 1024;
static constexpr int BS = B * S;            // 18432 rows
static constexpr int SD = S * D;            // 9216
static constexpr int SF = S * F;            // 403200

static constexpr int ACT_RELU_BF16    = 0;
static constexpr int ACT_NONE_F32     = 1;
static constexpr int ACT_SOFTPLUS_F32 = 2;

// ---------------- helpers ----------------
__device__ __forceinline__ uint16_t f2bf(float f) {
    uint32_t u = __float_as_uint(f);
    uint32_t r = u + 0x7FFFu + ((u >> 16) & 1u);   // round-to-nearest-even
    return (uint16_t)(r >> 16);
}
__device__ __forceinline__ uint32_t pack2bf(float lo, float hi) {
    return (uint32_t)f2bf(lo) | ((uint32_t)f2bf(hi) << 16);
}

// ---------------- fp32 -> bf16 elementwise ----------------
__global__ void cvt_f32_bf16(const float* __restrict__ in, uint16_t* __restrict__ out, int n) {
    int i = blockIdx.x * blockDim.x + threadIdx.x;
    int stride = gridDim.x * blockDim.x;
    for (; i < n; i += stride) out[i] = f2bf(in[i]);
}

// ---------------- WMMA GEMM: out = act(A[M,K](bf16) @ W[K,N](f32->bf16) + bias) ----------------
// block: nwaves*32 threads. grid.x covers N in nwaves*16 chunks, grid.y = M/16.
// A tile (16 x 64 bf16, uniform per workgroup) staged in LDS with 33-dword row pitch
// (conflict-free fragment reads). TDM path: D# descriptor with pad_interval=32dw/
// pad_amount=1dw produces that pitch directly; tensor_dim0 = K-k0 zero-fills the K tail.
__global__ void gemm_wmma_bf16(const uint16_t* __restrict__ A, int lda,
                               const float* __restrict__ W,
                               const float* __restrict__ bias,
                               int M, int N, int K,
                               void* __restrict__ out, int ldo, int act) {
    const int tid    = threadIdx.x;
    const int wave   = tid >> 5;
    const int lane   = tid & 31;
    const int half   = lane >> 4;       // 0: lanes 0-15, 1: lanes 16-31
    const int mrow   = lane & 15;       // A-fragment row
    const int ncol   = lane & 15;       // B/C/D fragment column
    const int nwaves = blockDim.x >> 5;

    const int row0 = blockIdx.y * 16;
    const int nt   = (blockIdx.x * nwaves + wave) * 16;   // this wave's N-tile start
    const bool active = (nt < N);
    const int n_g = nt + ncol;
    const bool nt_stream = (act == ACT_SOFTPLUS_F32);     // dw2: 825MB use-once stream

    __shared__ uint32_t lds_a32[16 * 33 + 16];            // 16 rows x (32 data + 1 pad) dwords

    union { v8f v; float f[8]; } acc;
    #pragma unroll
    for (int r = 0; r < 8; ++r) acc.f[r] = 0.0f;

    for (int k0 = 0; k0 < K; k0 += 64) {
#ifdef HAVE_TDM
        if (wave == 0) {
            // ---- Tensor Data Mover: DMA 16x64 bf16 tile (row stride lda) into LDS ----
            const uint32_t lds_off = (uint32_t)(uintptr_t)(void*)&lds_a32[0]; // low 32b = LDS offset
            const uint64_t gaddr   = (uint64_t)(uintptr_t)(A) +
                                     ((uint64_t)row0 * (uint64_t)lda + (uint64_t)k0) * 2ull;
            // D# group 0: count=1 | lds_addr | global_addr[56:0] | type=2
            v4u g0 = { 1u, lds_off, (uint32_t)gaddr,
                       (uint32_t)((gaddr >> 32) & 0x01FFFFFFull) | 0x80000000u };
            // D# group 1
            const uint32_t dim0 = (uint32_t)(K - k0);     // remaining K: HW zero-fills OOB tail
            const uint32_t w0 = (1u << 16)                 // data_size = 2 bytes
                              | (1u << 20)                 // pad_enable
                              | (4u << 22);                // pad_interval = 32 dwords (=1 row)
                                                           // pad_amount  = 1 dword (code 0)
            v8i g1 = { (int)w0,
                       (int)((dim0 & 0xFFFFu) << 16),      // tensor_dim0[15:0]
                       (int)((dim0 >> 16) | (16u << 16)),  // tensor_dim0[31:16] | tensor_dim1=16
                       (int)(64u << 16),                   // tile_dim0 = 64
                       (int)16,                            // tile_dim1 = 16, tile_dim2 = 0
                       (int)(uint32_t)lda,                 // tensor_dim0_stride[31:0]
                       0, 0 };
            v4i gz = { 0, 0, 0, 0 };
#if defined(__clang_major__) && (__clang_major__ >= 23)
            v8i gz8 = { 0, 0, 0, 0, 0, 0, 0, 0 };
            __builtin_amdgcn_tensor_load_to_lds(g0, g1, gz, gz, gz8, 0);
#else
            __builtin_amdgcn_tensor_load_to_lds(g0, g1, gz, gz, 0);
#endif
            __builtin_amdgcn_s_wait_tensorcnt(0);
        }
        __syncthreads();
#else
        // ---- fallback: cooperative dword loader (K always even, rows dword-aligned) ----
        for (int i = tid; i < 16 * 32; i += blockDim.x) {
            int m = i >> 5, dk = i & 31;
            int kg = k0 + 2 * dk;
            uint32_t v = 0;
            if (kg < K)
                v = *(const uint32_t*)(A + (size_t)(row0 + m) * lda + kg);
            lds_a32[m * 33 + dk] = v;
        }
        __syncthreads();
#endif
        if (active) {
            const uint32_t* arow = &lds_a32[mrow * 33];
            #pragma unroll
            for (int t = 0; t < 2; ++t) {                 // two 16x16x32 WMMAs per 64-K step
                union { v16bf v; uint32_t u[8]; } af;
                #pragma unroll
                for (int j = 0; j < 8; ++j) {             // 16-bit A 16x32 layout (ISA 7.12.2)
                    int ka = (j < 4) ? (half * 8 + 2 * j) : (16 + half * 8 + 2 * (j - 4));
                    af.u[j] = arow[t * 16 + (ka >> 1)];
                }
                union { v16bf v; uint32_t u[8]; } bfg;    // B 32x16: same K order as A (lanes 0-15 K=0..7,16..23; 16-31 K=8..15,24..31)
                #pragma unroll
                for (int j = 0; j < 8; ++j) {
                    int ka = k0 + t * 32 + ((j < 4) ? (half * 8 + 2 * j) : (16 + half * 8 + 2 * (j - 4)));
                    float w0 = 0.0f, w1 = 0.0f;
                    if (ka < K) {
                        const float* p0 = &W[(size_t)ka * N + n_g];
                        const float* p1 = &W[(size_t)(ka + 1) * N + n_g];
                        if (nt_stream) { w0 = __builtin_nontemporal_load(p0);
                                         w1 = __builtin_nontemporal_load(p1); }
                        else           { w0 = *p0; w1 = *p1; }
                    }
                    bfg.u[j] = pack2bf(w0, w1);
                }
                acc.v = __builtin_amdgcn_wmma_f32_16x16x32_bf16(
                            false, af.v, false, bfg.v, (short)0, acc.v, false, false);
            }
        }
        __syncthreads();
    }

    if (active) {
        const float bv = bias[n_g];
        #pragma unroll
        for (int r = 0; r < 8; ++r) {
            int m = row0 + r + half * 8;          // C/D layout: VGPR r -> M = r (+8 for hi half)
            float v = acc.f[r] + bv;
            if (act == ACT_RELU_BF16) {
                v = v > 0.0f ? v : 0.0f;
                ((uint16_t*)out)[(size_t)m * ldo + n_g] = f2bf(v);
            } else if (act == ACT_NONE_F32) {
                ((float*)out)[(size_t)m * ldo + n_g] = v;
            } else { // softplus
                float sp = (v > 20.0f) ? v : log1pf(expf(v));
                ((float*)out)[(size_t)m * ldo + n_g] = sp;
            }
        }
    }
}

// ---------------- split enc -> mean/log_var outputs, reparameterize z_e ----------------
__global__ void reparam_kernel(const float* __restrict__ enc,   // [BS, 32]
                               const float* __restrict__ eps,   // [BS*16]
                               float* __restrict__ mean_out,
                               float* __restrict__ lv_out,
                               float* __restrict__ z_e) {
    int i = blockIdx.x * blockDim.x + threadIdx.x;               // [0, BS*16)
    if (i >= BS * D) return;
    int r = i >> 4, d = i & 15;
    float m  = enc[(size_t)r * 32 + d];
    float lv = enc[(size_t)r * 32 + 16 + d];
    mean_out[i] = m;
    lv_out[i]   = lv;
    z_e[i]      = m + expf(0.5f * lv) * eps[i];
}

__global__ void zero_scalar(float* p) { *p = 0.0f; }

// ---------------- VQ: nearest codebook row (fp32), zf (bf16), vq_loss ----------------
__global__ void vq_kernel(const float* __restrict__ z_e,
                          const float* __restrict__ codebook,   // [1024,16]
                          uint16_t* __restrict__ zf,            // [BS*16] bf16
                          float* __restrict__ loss) {
    __shared__ float cb[K_CB * D];                               // 64 KB of 320 KB LDS
    for (int i = threadIdx.x; i < K_CB * D; i += blockDim.x) cb[i] = codebook[i];
    __syncthreads();

    int r = blockIdx.x * blockDim.x + threadIdx.x;               // 72*256 == 18432 exactly
    float z[D];
    #pragma unroll
    for (int d = 0; d < D; ++d) z[d] = z_e[(size_t)r * D + d];

    float best = 3.4e38f; int bi = 0;
    for (int k = 0; k < K_CB; ++k) {
        float dist = 0.0f;
        #pragma unroll
        for (int d = 0; d < D; ++d) {
            float diff = z[d] - cb[k * D + d];
            dist = fmaf(diff, diff, dist);
        }
        if (dist < best) { best = dist; bi = k; }                // first-min, matches argmin
    }

    float lsum = 0.0f;
    #pragma unroll
    for (int d = 0; d < D; ++d) {
        float q = cb[bi * D + d];
        zf[(size_t)r * D + d] = f2bf(q);
        float diff = z[d] - q;
        lsum = fmaf(diff, diff, lsum);
    }
    // wave32 reduction, one atomic per wave
    #pragma unroll
    for (int off = 16; off > 0; off >>= 1) lsum += __shfl_xor(lsum, off, 32);
    if ((threadIdx.x & 31) == 0)
        atomicAdd(loss, lsum * (1.0f / (float)(BS * D)));
}

// ---------------- host side ----------------
extern "C" void kernel_launch(void* const* d_in, const int* in_sizes, int n_in,
                              void* d_out, int out_size, void* d_ws, size_t ws_size,
                              hipStream_t stream) {
    (void)in_sizes; (void)n_in; (void)out_size; (void)ws_size;
    const float* x   = (const float*)d_in[0];
    const float* eps = (const float*)d_in[1];
    const float* ew0 = (const float*)d_in[2];  const float* eb0 = (const float*)d_in[3];
    const float* ew1 = (const float*)d_in[4];  const float* eb1 = (const float*)d_in[5];
    const float* ew2 = (const float*)d_in[6];  const float* eb2 = (const float*)d_in[7];
    const float* dw0 = (const float*)d_in[8];  const float* db0 = (const float*)d_in[9];
    const float* dw1 = (const float*)d_in[10]; const float* db1 = (const float*)d_in[11];
    const float* dw2 = (const float*)d_in[12]; const float* db2 = (const float*)d_in[13];
    const float* cbk = (const float*)d_in[14];

    // output regions: recon [B,S,F], mean [B,S,D], log_var [B,S,D], loss scalar
    float* recon    = (float*)d_out;
    float* mean_out = recon + (size_t)B * SF;
    float* lv_out   = mean_out + (size_t)BS * D;
    float* loss     = lv_out + (size_t)BS * D;

    // workspace packing (lifetime-aliased, peak 42.6 MB)
    char* ws = (char*)d_ws;
    uint16_t* x16  = (uint16_t*)(ws + 0);               //  25,804,800 B  [dies after G1]
    uint16_t* h1   = (uint16_t*)(ws + 25804800);        //  18,874,368 B
    uint16_t* h2   = (uint16_t*)(ws + 0);               //   9,437,184 B  (reuses x16)
    float*    encb = (float*)   (ws + 9437184);         //   2,359,296 B
    float*    ze   = (float*)   (ws + 11796480);        //   1,179,648 B
    uint16_t* zf   = (uint16_t*)(ws + 12976128);        //     589,824 B
    uint16_t* g1   = (uint16_t*)(ws + 13565952);        //      65,536 B
    uint16_t* g2   = (uint16_t*)(ws + 13631488);        //      32,768 B

    // x -> bf16 (one pass; A-side of encoder L1)
    cvt_f32_bf16<<<2048, 256, 0, stream>>>(x, x16, BS * F);

    // encoder (TimeDistributed dense over 18432 rows)
    gemm_wmma_bf16<<<dim3(4, BS / 16), 256, 0, stream>>>(x16, F,  ew0, eb0, BS, 512, F,   h1,  512, ACT_RELU_BF16);
    gemm_wmma_bf16<<<dim3(2, BS / 16), 256, 0, stream>>>(h1, 512, ew1, eb1, BS, 256, 512, h2,  256, ACT_RELU_BF16);
    gemm_wmma_bf16<<<dim3(1, BS / 16),  64, 0, stream>>>(h2, 256, ew2, eb2, BS,  32, 256, encb, 32, ACT_NONE_F32);

    // reparameterize + emit mean/log_var
    reparam_kernel<<<(BS * D) / 256, 256, 0, stream>>>(encb, eps, mean_out, lv_out, ze);

    // vector quantization (fp32 argmin, codebook in LDS) + vq_loss
    zero_scalar<<<1, 1, 0, stream>>>(loss);
    vq_kernel<<<BS / 256, 256, 0, stream>>>(ze, cbk, zf, loss);

    // decoder (M = 32 batch rows)
    gemm_wmma_bf16<<<dim3(8,    2), 256, 0, stream>>>(zf, SD,  dw0, db0, B, 1024, SD,   g1, 1024, ACT_RELU_BF16);
    gemm_wmma_bf16<<<dim3(4,    2), 256, 0, stream>>>(g1, 1024, dw1, db1, B,  512, 1024, g2,  512, ACT_RELU_BF16);
    gemm_wmma_bf16<<<dim3(3150, 2), 256, 0, stream>>>(g2,  512, dw2, db2, B,   SF,  512, recon, SF, ACT_SOFTPLUS_F32);
}
